// Llama4MoE_42691974922803
// MI455X (gfx1250) — compile-verified, matrix operand order checked by reading, not run on hardware
//
#include <hip/hip_runtime.h>
#include <hip/hip_bf16.h>

typedef __attribute__((ext_vector_type(16))) __bf16 v16bf;
typedef __attribute__((ext_vector_type(8)))  __bf16 v8bf;
typedef __attribute__((ext_vector_type(4)))  __bf16 v4bf;
typedef __attribute__((ext_vector_type(8)))  float  v8f;
typedef __attribute__((ext_vector_type(4)))  float  v4f;

#define TT   4096
#define DD   1024
#define II   2048
#define EE   16
#define BM   64
#define BN   128
#define BK   32
#define LPAD 8
#define RPAD (TT + EE * BM)   /* 5120 padded gathered rows */
#define MAXT 96               /* worst case sum ceil(c_e/64) = 79 */
#define ZROWI 544             /* ints in zero-row: (1024+32)*2B / 4 rounded up */

__device__ __forceinline__ float fast_sigmoid(float x) {
  return __builtin_amdgcn_rcpf(1.f + __expf(-x));   // v_rcp_f32, no div chain
}

// ---------------------------------------------------------------- init
__global__ void moe_init(int* gidx, int* counts, int* fill, int* zrow) {
  int i = blockIdx.x * 256 + threadIdx.x;
  if (i < RPAD) gidx[i] = -1;
  if (i < EE) { counts[i] = 0; fill[i] = 0; }
  if (i < ZROWI) zrow[i] = 0;
}

// ---------------------------------------------------------------- router
__global__ __launch_bounds__(128)
void moe_router(const float* __restrict__ hidden, const float* __restrict__ w_router,
                __bf16* __restrict__ xs, __bf16* __restrict__ hs,
                int* __restrict__ topIdx, int* __restrict__ counts) {
  int t = blockIdx.x, tid = threadIdx.x;
  const float* hrow = hidden + (long)t * DD;

  // each thread owns 8 contiguous elements of D (128*8 = 1024)
  int d0 = tid * 8;
  float x[8];
#pragma unroll
  for (int j = 0; j < 8; ++j) x[j] = hrow[d0 + j];

  float part[EE];
#pragma unroll
  for (int e = 0; e < EE; ++e) part[e] = 0.f;
#pragma unroll
  for (int j = 0; j < 8; ++j) {
    const float* wr = w_router + (long)(d0 + j) * EE;
#pragma unroll
    for (int e = 0; e < EE; ++e) part[e] += x[j] * wr[e];
  }

  __shared__ float red[EE][128];
  __shared__ float sScore;
#pragma unroll
  for (int e = 0; e < EE; ++e) red[e][tid] = part[e];
  __syncthreads();

  if (tid < EE) {
    float s = 0.f;
    for (int i = 0; i < 128; ++i) s += red[tid][i];
    red[tid][0] = s;
  }
  __syncthreads();

  if (tid == 0) {
    int best = 0; float bl = red[0][0];
#pragma unroll
    for (int e = 1; e < EE; ++e)
      if (red[e][0] > bl) { bl = red[e][0]; best = e; }
    topIdx[t] = best;
    sScore = fast_sigmoid(bl);
    atomicAdd(&counts[best], 1);
  }
  __syncthreads();

  float sc = sScore;
  v8bf vx, vs;
#pragma unroll
  for (int j = 0; j < 8; ++j) {
    vs[j] = (__bf16)(x[j] * sc);
    vx[j] = (__bf16)x[j];
  }
  *(v8bf*)(xs + (long)t * DD + d0) = vs;   // single b128 store
  *(v8bf*)(hs + (long)t * DD + d0) = vx;   // single b128 store
}

// ---------------------------------------------------------------- scan (1 thread)
__global__ void moe_scan(const int* __restrict__ counts, int* poff,
                         int* tileE, int* tileR, int* fill) {
  if (threadIdx.x != 0) return;
  int off = 0, tile = 0;
  for (int e = 0; e < EE; ++e) {
    poff[e] = off;
    fill[e] = 0;
    int nt = (counts[e] + BM - 1) / BM;
    for (int i = 0; i < nt && tile < MAXT; ++i) {
      tileE[tile] = e; tileR[tile] = off + i * BM; ++tile;
    }
    off += nt * BM;
  }
  poff[EE] = off;
  for (int i = tile; i < MAXT; ++i) tileE[i] = -1;
}

// ---------------------------------------------------------------- scatter
__global__ void moe_scatter(const int* __restrict__ topIdx, const int* __restrict__ poff,
                            int* fill, int* gidx, int* gslot) {
  int t = blockIdx.x * 256 + threadIdx.x;
  if (t >= TT) return;
  int e = topIdx[t];
  int slot = atomicAdd(&fill[e], 1);
  int row = poff[e] + slot;
  gidx[row] = t;
  gslot[t] = row;
}

// ---------------------------------------------------------------- fused gate/up GEMM + silu
__global__ __launch_bounds__(256)
void moe_gemm_gateup(const __bf16* __restrict__ A, const int* __restrict__ gidx,
                     const int* __restrict__ tileExpert, const int* __restrict__ tileRow,
                     const float* __restrict__ WgB, const float* __restrict__ WuB,
                     __bf16* __restrict__ H, const __bf16* __restrict__ zrow,
                     int K, int N, long wstride) {
  __shared__ __bf16 As[BM][BK + LPAD];
  __shared__ __bf16 Bg[BN][BK + LPAD];
  __shared__ __bf16 Bu[BN][BK + LPAD];

  int bm = blockIdx.x, bn = blockIdx.y;
  int expert = 0, rowBase;
  if (tileExpert) {
    expert = tileExpert[bm];
    if (expert < 0) return;
    rowBase = tileRow[bm];
  } else rowBase = bm * BM;

  int tid = threadIdx.x;
  int ar = tid >> 2;
  int ac = (tid & 3) * 8;
  long atok = rowBase + ar;
  if (gidx) atok = gidx[rowBase + ar];
  // padded/invalid rows read from a zeroed scratch row: no branch in hot loop
  const __bf16* Aptr = (atok >= 0) ? (A + atok * (long)K + ac) : (zrow + ac);
  long astep = (atok >= 0) ? BK : 0;

  int bnc = (tid & 31) * 4;
  int bkr = (tid >> 5) * 4;
  const long wrow = (long)N;            // elements per weight row
  const long wtile = (long)BK * N;      // elements per K-tile of weights
  const float* wg0 = WgB + (long)expert * wstride + (long)bkr * N + (long)bn * BN + bnc;
  const float* wu0 = WuB + (long)expert * wstride + (long)bkr * N + (long)bn * BN + bnc;

  int wid = tid >> 5, lane = tid & 31;
  int wm = wid >> 2, wn = wid & 3;
  int lm = lane & 15, kh = lane >> 4;

  v8f accG[2][2] = {};
  v8f accU[2][2] = {};
  union V16 { v16bf v; v8bf h[2]; };

  for (int k0 = 0; k0 < K; k0 += BK) {
    *(v8bf*)&As[ar][ac] = *(const v8bf*)Aptr;
    Aptr += astep;

    v4f g0 = *(const v4f*)(wg0);
    v4f g1 = *(const v4f*)(wg0 + wrow);
    v4f g2 = *(const v4f*)(wg0 + 2 * wrow);
    v4f g3 = *(const v4f*)(wg0 + 3 * wrow);
    v4f u0 = *(const v4f*)(wu0);
    v4f u1 = *(const v4f*)(wu0 + wrow);
    v4f u2 = *(const v4f*)(wu0 + 2 * wrow);
    v4f u3 = *(const v4f*)(wu0 + 3 * wrow);

    // prefetch next K-tile's weight rows (gfx1250 global_prefetch_b8)
    if (k0 + BK < K) {
      __builtin_prefetch(wg0 + wtile, 0, 1);
      __builtin_prefetch(wu0 + wtile, 0, 1);
    }
    wg0 += wtile;
    wu0 += wtile;

#pragma unroll
    for (int j = 0; j < 4; ++j) {
      v4bf tg; tg[0] = (__bf16)g0[j]; tg[1] = (__bf16)g1[j]; tg[2] = (__bf16)g2[j]; tg[3] = (__bf16)g3[j];
      *(v4bf*)&Bg[bnc + j][bkr] = tg;
      v4bf tu; tu[0] = (__bf16)u0[j]; tu[1] = (__bf16)u1[j]; tu[2] = (__bf16)u2[j]; tu[3] = (__bf16)u3[j];
      *(v4bf*)&Bu[bnc + j][bkr] = tu;
    }
    __syncthreads();

    v16bf af[2], bgf[2], buf[2];
#pragma unroll
    for (int f = 0; f < 2; ++f) {
      V16 t;
      t.h[0] = *(const v8bf*)&As[wm * 32 + f * 16 + lm][kh * 8];
      t.h[1] = *(const v8bf*)&As[wm * 32 + f * 16 + lm][16 + kh * 8];
      af[f] = t.v;
      t.h[0] = *(const v8bf*)&Bg[wn * 32 + f * 16 + lm][kh * 8];
      t.h[1] = *(const v8bf*)&Bg[wn * 32 + f * 16 + lm][16 + kh * 8];
      bgf[f] = t.v;
      t.h[0] = *(const v8bf*)&Bu[wn * 32 + f * 16 + lm][kh * 8];
      t.h[1] = *(const v8bf*)&Bu[wn * 32 + f * 16 + lm][16 + kh * 8];
      buf[f] = t.v;
    }
#pragma unroll
    for (int fm = 0; fm < 2; ++fm)
#pragma unroll
      for (int fn = 0; fn < 2; ++fn) {
        accG[fm][fn] = __builtin_amdgcn_wmma_f32_16x16x32_bf16(
            false, af[fm], false, bgf[fn], (short)0, accG[fm][fn], false, false);
        accU[fm][fn] = __builtin_amdgcn_wmma_f32_16x16x32_bf16(
            false, af[fm], false, buf[fn], (short)0, accU[fm][fn], false, false);
      }
    __syncthreads();
  }

#pragma unroll
  for (int fm = 0; fm < 2; ++fm)
#pragma unroll
    for (int fn = 0; fn < 2; ++fn) {
      int col = bn * BN + wn * 32 + fn * 16 + lm;
      long row0 = (long)rowBase + wm * 32 + fm * 16 + kh * 8;
#pragma unroll
      for (int r = 0; r < 8; ++r) {
        float g = accG[fm][fn][r];
        float u = accU[fm][fn][r];
        float h = g * fast_sigmoid(g) * u;
        H[(row0 + r) * (long)N + col] = (__bf16)h;
      }
    }
}

// ---------------------------------------------------------------- down GEMM (f32 out)
__global__ __launch_bounds__(256)
void moe_gemm_down(const __bf16* __restrict__ A,
                   const int* __restrict__ tileExpert, const int* __restrict__ tileRow,
                   const float* __restrict__ WdB,
                   float* __restrict__ Out, int K, int N, long wstride) {
  __shared__ __bf16 As[BM][BK + LPAD];
  __shared__ __bf16 Bs[BN][BK + LPAD];

  int bm = blockIdx.x, bn = blockIdx.y;
  int expert = 0, rowBase;
  if (tileExpert) {
    expert = tileExpert[bm];
    if (expert < 0) return;
    rowBase = tileRow[bm];
  } else rowBase = bm * BM;

  int tid = threadIdx.x;
  int ar = tid >> 2;
  int ac = (tid & 3) * 8;
  const __bf16* Aptr = A + ((long)rowBase + ar) * K + ac;
  int bnc = (tid & 31) * 4;
  int bkr = (tid >> 5) * 4;
  const long wrow = (long)N;
  const long wtile = (long)BK * N;
  const float* wd0 = WdB + (long)expert * wstride + (long)bkr * N + (long)bn * BN + bnc;

  int wid = tid >> 5, lane = tid & 31;
  int wm = wid >> 2, wn = wid & 3;
  int lm = lane & 15, kh = lane >> 4;

  v8f acc[2][2] = {};
  union V16 { v16bf v; v8bf h[2]; };

  for (int k0 = 0; k0 < K; k0 += BK) {
    *(v8bf*)&As[ar][ac] = *(const v8bf*)Aptr;
    Aptr += BK;

    v4f d0 = *(const v4f*)(wd0);
    v4f d1 = *(const v4f*)(wd0 + wrow);
    v4f d2 = *(const v4f*)(wd0 + 2 * wrow);
    v4f d3 = *(const v4f*)(wd0 + 3 * wrow);

    if (k0 + BK < K)
      __builtin_prefetch(wd0 + wtile, 0, 1);
    wd0 += wtile;

#pragma unroll
    for (int j = 0; j < 4; ++j) {
      v4bf td; td[0] = (__bf16)d0[j]; td[1] = (__bf16)d1[j]; td[2] = (__bf16)d2[j]; td[3] = (__bf16)d3[j];
      *(v4bf*)&Bs[bnc + j][bkr] = td;
    }
    __syncthreads();

    v16bf af[2], bf[2];
#pragma unroll
    for (int f = 0; f < 2; ++f) {
      V16 t;
      t.h[0] = *(const v8bf*)&As[wm * 32 + f * 16 + lm][kh * 8];
      t.h[1] = *(const v8bf*)&As[wm * 32 + f * 16 + lm][16 + kh * 8];
      af[f] = t.v;
      t.h[0] = *(const v8bf*)&Bs[wn * 32 + f * 16 + lm][kh * 8];
      t.h[1] = *(const v8bf*)&Bs[wn * 32 + f * 16 + lm][16 + kh * 8];
      bf[f] = t.v;
    }
#pragma unroll
    for (int fm = 0; fm < 2; ++fm)
#pragma unroll
      for (int fn = 0; fn < 2; ++fn)
        acc[fm][fn] = __builtin_amdgcn_wmma_f32_16x16x32_bf16(
            false, af[fm], false, bf[fn], (short)0, acc[fm][fn], false, false);
    __syncthreads();
  }

#pragma unroll
  for (int fm = 0; fm < 2; ++fm)
#pragma unroll
    for (int fn = 0; fn < 2; ++fn) {
      int col = bn * BN + wn * 32 + fn * 16 + lm;
      long row0 = (long)rowBase + wm * 32 + fm * 16 + kh * 8;
#pragma unroll
      for (int r = 0; r < 8; ++r)
        Out[(row0 + r) * (long)N + col] = acc[fm][fn][r];
    }
}

// ---------------------------------------------------------------- combine (x4 vectorized)
__global__ void moe_combine(float* __restrict__ out, const float* __restrict__ ro,
                            const int* __restrict__ gslot) {
  long i = ((long)blockIdx.x * 256 + threadIdx.x) * 4;
  int t = (int)(i >> 10);
  int d = (int)(i & (DD - 1));
  v4f a = *(const v4f*)(out + i);
  v4f b = *(const v4f*)(ro + (long)gslot[t] * DD + d);
  *(v4f*)(out + i) = a + b;
}

// ---------------------------------------------------------------- launch
extern "C" void kernel_launch(void* const* d_in, const int* in_sizes, int n_in,
                              void* d_out, int out_size, void* d_ws, size_t ws_size,
                              hipStream_t stream) {
  const float* hidden   = (const float*)d_in[0];
  const float* w_router = (const float*)d_in[1];
  const float* w_gate   = (const float*)d_in[2];
  const float* w_up     = (const float*)d_in[3];
  const float* w_down   = (const float*)d_in[4];
  const float* ws_gate  = (const float*)d_in[5];
  const float* ws_up    = (const float*)d_in[6];
  const float* ws_down  = (const float*)d_in[7];
  float* out = (float*)d_out;

  char* ws = (char*)d_ws;
  size_t off = 0;
  auto take = [&](size_t b) { size_t o = off; off += (b + 255) & ~(size_t)255; return o; };

  __bf16* xs   = (__bf16*)(ws + take((size_t)TT * DD * 2));
  __bf16* hs   = (__bf16*)(ws + take((size_t)TT * DD * 2));
  __bf16* Hr   = (__bf16*)(ws + take((size_t)RPAD * II * 2));
  __bf16* Hsh  = (__bf16*)(ws + take((size_t)TT * II * 2));
  float*  Ro   = (float*)(ws + take((size_t)RPAD * DD * 4));
  int* counts  = (int*)(ws + take(EE * 4));
  int* fill    = (int*)(ws + take(EE * 4));
  int* topIdx  = (int*)(ws + take((size_t)TT * 4));
  int* gslot   = (int*)(ws + take((size_t)TT * 4));
  int* gidx    = (int*)(ws + take((size_t)RPAD * 4));
  int* poff    = (int*)(ws + take((EE + 1) * 4));
  int* tileE   = (int*)(ws + take(MAXT * 4));
  int* tileR   = (int*)(ws + take(MAXT * 4));
  int* zrow    = (int*)(ws + take(ZROWI * 4));

  moe_init<<<(RPAD + 255) / 256, 256, 0, stream>>>(gidx, counts, fill, zrow);
  moe_router<<<TT, 128, 0, stream>>>(hidden, w_router, xs, hs, topIdx, counts);
  moe_scan<<<1, 32, 0, stream>>>(counts, poff, tileE, tileR, fill);
  moe_scatter<<<(TT + 255) / 256, 256, 0, stream>>>(topIdx, poff, fill, gidx, gslot);

  dim3 g1r(MAXT, II / BN);
  moe_gemm_gateup<<<g1r, 256, 0, stream>>>(xs, gidx, tileE, tileR, w_gate, w_up,
                                           Hr, (const __bf16*)zrow, DD, II, (long)DD * II);
  dim3 g1s(TT / BM, II / BN);
  moe_gemm_gateup<<<g1s, 256, 0, stream>>>(hs, nullptr, nullptr, nullptr, ws_gate, ws_up,
                                           Hsh, (const __bf16*)zrow, DD, II, 0);
  dim3 g2r(MAXT, DD / BN);
  moe_gemm_down<<<g2r, 256, 0, stream>>>(Hr, tileE, tileR, w_down, Ro, II, DD, (long)II * DD);
  dim3 g2s(TT / BM, DD / BN);
  moe_gemm_down<<<g2s, 256, 0, stream>>>(Hsh, nullptr, nullptr, ws_down, out, II, DD, 0);

  moe_combine<<<(TT * DD) / 1024, 256, 0, stream>>>(out, Ro, gslot);
}
